// SelfAttentionModule_23776938951309
// MI455X (gfx1250) — matrix operand order repaired and checked by reading, not compiled, not run on hardware
//
#include <hip/hip_runtime.h>
#include <cstdint>

typedef __attribute__((ext_vector_type(16))) _Float16 v16h;
typedef __attribute__((ext_vector_type(8)))  _Float16 v8h;
typedef __attribute__((ext_vector_type(8)))  float    v8f;
typedef __attribute__((ext_vector_type(4)))  float    v4f;

#define CCH   128
#define MTOT  32768      // 64*64*8 tokens
#define NQKV  384

// ---------------------------------------------------------------------------
// CDNA5 async global->LDS copy (16B per lane), tracked by ASYNCcnt.
// ldsOff: LDS byte address (generic shared ptr truncated to 32 bits).
// ---------------------------------------------------------------------------
__device__ inline void async_copy16(unsigned ldsOff, const void* gptr) {
  unsigned long long ga = (unsigned long long)(uintptr_t)gptr;
  asm volatile("global_load_async_to_lds_b128 %0, %1, off"
               :: "v"(ldsOff), "v"(ga) : "memory");
}
__device__ inline void wait_async0() {
  asm volatile("s_wait_asynccnt 0" ::: "memory");
}

__device__ inline v8f wmma16(v16h a, v16h b, v8f c) {
  return __builtin_amdgcn_wmma_f32_16x16x32_f16(false, a, false, b, (short)0, c, false, false);
}

// A 16x32 f16 fragment: lane<16 row=lane, halves = K{k0..k0+7, k0+16..k0+23},
// k0 = +8 for lanes>=16. Two 16B chunks 32B apart.
__device__ inline v16h ldA(const _Float16* p) {
  v8h lo = *(const v8h*)p;
  v8h hi = *(const v8h*)(p + 16);
  return __builtin_shufflevector(lo, hi, 0,1,2,3,4,5,6,7,8,9,10,11,12,13,14,15);
}

// ---------------------------------------------------------------------------
// 1) per-channel mean + rsqrt(var+eps)   (instance norm stats)
// ---------------------------------------------------------------------------
__global__ void stats_k(const float* __restrict__ x, float* __restrict__ muRs) {
  __shared__ float s1[256], s2[256];
  int c = blockIdx.x;
  const float* p = x + (size_t)c * MTOT;
  float s = 0.f, q = 0.f;
  for (int i = threadIdx.x; i < MTOT; i += 256) { float v = p[i]; s += v; q += v * v; }
  s1[threadIdx.x] = s; s2[threadIdx.x] = q;
  __syncthreads();
  for (int o = 128; o > 0; o >>= 1) {
    if (threadIdx.x < o) { s1[threadIdx.x] += s1[threadIdx.x + o]; s2[threadIdx.x] += s2[threadIdx.x + o]; }
    __syncthreads();
  }
  if (threadIdx.x == 0) {
    float mu  = s1[0] * (1.0f / MTOT);
    float var = s2[0] * (1.0f / MTOT) - mu * mu;
    muRs[c]       = mu;
    muRs[CCH + c] = rsqrtf(var + 1e-5f);
  }
}

// ---------------------------------------------------------------------------
// 2) weight conversion f32 -> f16
// ---------------------------------------------------------------------------
__global__ void wconv_k(const float* __restrict__ qw, const float* __restrict__ pw,
                        _Float16* __restrict__ wh, _Float16* __restrict__ ph) {
  int i = blockIdx.x * 256 + threadIdx.x;
  if (i < 49152) wh[i] = (_Float16)qw[i];
  else           ph[i - 49152] = (_Float16)pw[i - 49152];
}

// ---------------------------------------------------------------------------
// 3) normalize + transpose (C, M) -> (M, C) in f16 via LDS 32x32 tiles
// ---------------------------------------------------------------------------
__global__ void xpose_k(const float* __restrict__ x, const float* __restrict__ muRs,
                        _Float16* __restrict__ xh) {
  __shared__ float t[32][33];
  int mBase = blockIdx.x * 32, cBase = blockIdx.y * 32;
  #pragma unroll
  for (int r = 0; r < 4; r++) {
    int cl = threadIdx.y + r * 8;
    int c  = cBase + cl;
    float v = x[(size_t)c * MTOT + mBase + threadIdx.x];
    t[cl][threadIdx.x] = (v - muRs[c]) * muRs[CCH + c];
  }
  __syncthreads();
  #pragma unroll
  for (int r = 0; r < 4; r++) {
    int ml = threadIdx.y + r * 8;
    xh[(size_t)(mBase + ml) * CCH + cBase + threadIdx.x] = (_Float16)t[threadIdx.x][ml];
  }
}

// ---------------------------------------------------------------------------
// 4) QKV GEMM: (32768 x 128) @ (384 x 128)^T + bias -> (32768 x 384) f16
//    Block: 32 M-rows x 128 N-strip. A tile staged via async global->LDS.
//    Each wave: one 16-wide N tile, two M sub-tiles (B frag reused, 8 WMMAs).
// ---------------------------------------------------------------------------
__global__ void __launch_bounds__(256) qkv_gemm_k(
    const _Float16* __restrict__ xh, const _Float16* __restrict__ wh,
    const float* __restrict__ bias, _Float16* __restrict__ qkvh) {
  __shared__ _Float16 atile[32 * 128];   // 8 KB A staging
  __shared__ _Float16 otile[32 * 128];   // 8 KB result staging
  int tid = threadIdx.x, wave = tid >> 5, lane = tid & 31;
  int Mtile = blockIdx.x * 32;
  int nb0   = blockIdx.y * 128;

  // --- async stage A tile (32 x 128 f16 = 8192 B = 512 x 16B chunks) ---
  {
    unsigned abase = (unsigned)(uintptr_t)atile;
    const _Float16* g = xh + (size_t)Mtile * CCH;
    async_copy16(abase + (unsigned)tid * 16u,         g + tid * 8);
    async_copy16(abase + (unsigned)(tid + 256) * 16u, g + (tid + 256) * 8);
  }
  wait_async0();
  __syncthreads();

  int n = nb0 + wave * 16 + (lane & 15);
  const _Float16* bp = wh + (size_t)n * CCH + ((lane < 16) ? 0 : 8);
  const _Float16* arow = atile + (size_t)(lane & 15) * CCH + ((lane >= 16) ? 8 : 0);
  v8f acc0 = {}, acc1 = {};
  #pragma unroll
  for (int kt = 0; kt < 4; kt++) {
    v16h b  = ldA(bp + kt * 32);
    v16h a0 = ldA(arow + kt * 32);             // rows 0..15   (ds_load)
    v16h a1 = ldA(arow + 16 * CCH + kt * 32);  // rows 16..31
    acc0 = wmma16(a0, b, acc0);
    acc1 = wmma16(a1, b, acc1);
  }
  float bf = bias[n];
  int col = wave * 16 + (lane & 15);
  int rb  = (lane >= 16) ? 8 : 0;
  #pragma unroll
  for (int i = 0; i < 8; i++) {
    otile[(rb + i) * 128 + col]        = (_Float16)(acc0[i] + bf);
    otile[(16 + rb + i) * 128 + col]   = (_Float16)(acc1[i] + bf);
  }
  __syncthreads();
  #pragma unroll
  for (int it = 0; it < 2; it++) {
    int r = (tid >> 4) + it * 16, cc = (tid & 15) * 8;
    *(uint4*)(qkvh + (size_t)(Mtile + r) * NQKV + nb0 + cc) =
        *(const uint4*)(&otile[r * 128 + cc]);
  }
}

// ---------------------------------------------------------------------------
// 5) 3D neighborhood attention (27 keys, 8 heads, head_dim 16) — VALU kernel
// ---------------------------------------------------------------------------
__global__ void attn_k(const _Float16* __restrict__ qkvh, const float* __restrict__ rpb,
                       _Float16* __restrict__ ao) {
  int row = blockIdx.x * 256 + threadIdx.x;
  int h = row & 7, m = row >> 3;
  int z = m & 7, y = (m >> 3) & 63, x = m >> 9;
  int sx = x - 1; sx = sx < 0 ? 0 : (sx > 61 ? 61 : sx);
  int sy = y - 1; sy = sy < 0 ? 0 : (sy > 61 ? 61 : sy);
  int sz = z - 1; sz = sz < 0 ? 0 : (sz > 5  ? 5  : sz);

  float q[16];
  const _Float16* qp = qkvh + (size_t)m * NQKV + h * 16;
  { v8h a = *(const v8h*)qp, b = *(const v8h*)(qp + 8);
    #pragma unroll
    for (int i = 0; i < 8; i++) { q[i] = (float)a[i] * 0.25f; q[i + 8] = (float)b[i] * 0.25f; } }

  float logits[27], mx = -1e30f;
  const float* rp = rpb + h * 125;
  int j = 0;
  for (int ox = 0; ox < 3; ox++)
   for (int oy = 0; oy < 3; oy++)
    for (int oz = 0; oz < 3; oz++, j++) {
      int km = (sx + ox) * 512 + (sy + oy) * 8 + (sz + oz);
      const _Float16* kp = qkvh + (size_t)km * NQKV + 128 + h * 16;
      v8h a = *(const v8h*)kp, b = *(const v8h*)(kp + 8);
      float d = 0.f;
      #pragma unroll
      for (int i = 0; i < 8; i++) d += q[i] * (float)a[i] + q[i + 8] * (float)b[i];
      d += rp[(sx - x + ox + 2) * 25 + (sy - y + oy + 2) * 5 + (sz - z + oz + 2)];
      logits[j] = d; mx = d > mx ? d : mx;
    }

  float accv[16];
  #pragma unroll
  for (int i = 0; i < 16; i++) accv[i] = 0.f;
  float den = 0.f;
  j = 0;
  for (int ox = 0; ox < 3; ox++)
   for (int oy = 0; oy < 3; oy++)
    for (int oz = 0; oz < 3; oz++, j++) {
      float p = __expf(logits[j] - mx); den += p;
      int km = (sx + ox) * 512 + (sy + oy) * 8 + (sz + oz);
      const _Float16* vp = qkvh + (size_t)km * NQKV + 256 + h * 16;
      v8h a = *(const v8h*)vp, b = *(const v8h*)(vp + 8);
      #pragma unroll
      for (int i = 0; i < 8; i++) { accv[i] += p * (float)a[i]; accv[i + 8] += p * (float)b[i]; }
    }
  float inv = 1.0f / den;
  v8h o0, o1;
  #pragma unroll
  for (int i = 0; i < 8; i++) { o0[i] = (_Float16)(accv[i] * inv); o1[i] = (_Float16)(accv[i + 8] * inv); }
  _Float16* op = ao + (size_t)m * CCH + h * 16;
  *(v8h*)op = o0; *(v8h*)(op + 8) = o1;
}

// ---------------------------------------------------------------------------
// 6) Output projection GEMM + bias, transposed f32 store to (C, H, W, Z).
//    Same async-A-staging, 2 M sub-tiles per wave, NT stores for output.
// ---------------------------------------------------------------------------
__global__ void __launch_bounds__(256) proj_gemm_k(
    const _Float16* __restrict__ ao, const _Float16* __restrict__ ph,
    const float* __restrict__ bias, float* __restrict__ out) {
  __shared__ _Float16 atile[32 * 128];   // 8 KB
  __shared__ float    otile[32 * 129];   // 16.5 KB, padded rows (M x N)
  int tid = threadIdx.x, wave = tid >> 5, lane = tid & 31;
  int Mtile = blockIdx.x * 32;

  {
    unsigned abase = (unsigned)(uintptr_t)atile;
    const _Float16* g = ao + (size_t)Mtile * CCH;
    async_copy16(abase + (unsigned)tid * 16u,         g + tid * 8);
    async_copy16(abase + (unsigned)(tid + 256) * 16u, g + (tid + 256) * 8);
  }
  wait_async0();
  __syncthreads();

  int n = wave * 16 + (lane & 15);
  const _Float16* bp = ph + (size_t)n * CCH + ((lane < 16) ? 0 : 8);
  const _Float16* arow = atile + (size_t)(lane & 15) * CCH + ((lane >= 16) ? 8 : 0);
  v8f acc0 = {}, acc1 = {};
  #pragma unroll
  for (int kt = 0; kt < 4; kt++) {
    v16h b  = ldA(bp + kt * 32);
    v16h a0 = ldA(arow + kt * 32);
    v16h a1 = ldA(arow + 16 * CCH + kt * 32);
    acc0 = wmma16(a0, b, acc0);
    acc1 = wmma16(a1, b, acc1);
  }
  float bf = bias[n];
  int rb = (lane >= 16) ? 8 : 0;
  #pragma unroll
  for (int i = 0; i < 8; i++) {
    otile[(rb + i) * 129 + n]        = acc0[i] + bf;
    otile[(16 + rb + i) * 129 + n]   = acc1[i] + bf;
  }
  __syncthreads();
  // transposed store: out[n][Mtile + m], 16 contiguous m per thread
  int nn = tid >> 1, mo = (tid & 1) * 16;
  float* dst = out + (size_t)nn * MTOT + Mtile + mo;
  #pragma unroll
  for (int g4 = 0; g4 < 4; g4++) {
    v4f v;
    v.x = otile[(mo + g4 * 4 + 0) * 129 + nn];
    v.y = otile[(mo + g4 * 4 + 1) * 129 + nn];
    v.z = otile[(mo + g4 * 4 + 2) * 129 + nn];
    v.w = otile[(mo + g4 * 4 + 3) * 129 + nn];
    __builtin_nontemporal_store(v, (v4f*)(dst + g4 * 4));
  }
}

// ---------------------------------------------------------------------------
extern "C" void kernel_launch(void* const* d_in, const int* in_sizes, int n_in,
                              void* d_out, int out_size, void* d_ws, size_t ws_size,
                              hipStream_t stream) {
  const float* x      = (const float*)d_in[0];
  const float* qkv_w  = (const float*)d_in[1];
  const float* qkv_b  = (const float*)d_in[2];
  const float* rpb    = (const float*)d_in[3];
  const float* proj_w = (const float*)d_in[4];
  const float* proj_b = (const float*)d_in[5];

  char* ws = (char*)d_ws;
  float*    muRs = (float*)ws;                                   // 1 KB
  _Float16* xh   = (_Float16*)(ws + 1024);                       // 8 MB  (M x 128)
  _Float16* wh   = (_Float16*)(ws + 1024 + 8388608);             // 96 KB (384 x 128)
  _Float16* ph   = (_Float16*)(ws + 1024 + 8388608 + 98304);     // 32 KB (128 x 128)
  _Float16* qkvh = (_Float16*)(ws + 1024 + 8388608 + 98304 + 32768);            // 24 MB (M x 384)
  _Float16* ao   = (_Float16*)(ws + 1024 + 8388608 + 98304 + 32768 + 25165824); // 8 MB  (M x 128)

  stats_k<<<128, 256, 0, stream>>>(x, muRs);
  wconv_k<<<256, 256, 0, stream>>>(qkv_w, proj_w, wh, ph);
  xpose_k<<<dim3(MTOT / 32, CCH / 32), dim3(32, 8), 0, stream>>>(x, muRs, xh);
  qkv_gemm_k<<<dim3(MTOT / 32, 3), 256, 0, stream>>>(xh, wh, qkv_b, qkvh);
  attn_k<<<(MTOT * 8) / 256, 256, 0, stream>>>(qkvh, rpb, ao);
  proj_gemm_k<<<MTOT / 32, 256, 0, stream>>>(ao, ph, proj_b, (float*)d_out);
}
